// DeformConv2d_84619445666681
// MI455X (gfx1250) — matrix operand order repaired and checked by reading, not compiled, not run on hardware
//
#include <hip/hip_runtime.h>
#include <hip/hip_bf16.h>
#include <stdint.h>

// ---------------- problem constants ----------------
#define Bn   8
#define Cc   256
#define Oc   256
#define Hh   56
#define Wd   56
#define HW   3136          // 56*56
#define KKp  9
#define NKT  72            // (KKp*Cc)/32 K-chunks of 32

typedef __attribute__((ext_vector_type(16))) __bf16 v16bf;
typedef __attribute__((ext_vector_type(8)))  __bf16 v8bf;
typedef __attribute__((ext_vector_type(8)))  float  v8f;

__device__ __forceinline__ unsigned int pk_bf16(float lo, float hi) {
  unsigned int ul = __float_as_uint(lo);
  ul = (ul + 0x7FFFu + ((ul >> 16) & 1u)) >> 16;
  unsigned int uh = __float_as_uint(hi);
  uh = (uh + 0x7FFFu + ((uh >> 16) & 1u)) & 0xFFFF0000u;
  return ul | uh;
}

// ------------- 1) NCHW -> NHWC transpose of x -------------
__global__ __launch_bounds__(256) void transpose_kernel(
    const float* __restrict__ x, float* __restrict__ xT) {
  __shared__ float t[32][33];
  const int hw0 = blockIdx.x * 32;
  const int c0  = blockIdx.y * 32;
  const int b   = blockIdx.z;
  const int tx = threadIdx.x, ty = threadIdx.y;
#pragma unroll
  for (int j = 0; j < 4; ++j)
    t[ty + j * 8][tx] = x[((size_t)(b * Cc + c0 + ty + j * 8)) * HW + hw0 + tx];
  __syncthreads();
#pragma unroll
  for (int j = 0; j < 4; ++j)
    xT[((size_t)b * HW + hw0 + ty + j * 8) * Cc + c0 + tx] = t[tx][ty + j * 8];
}

// ------------- 2) offset conv -> sampling coords py/px -------------
__global__ __launch_bounds__(224) void off_conv_kernel(
    const float* __restrict__ x, const float* __restrict__ w_off,
    const float* __restrict__ b_off, float* __restrict__ py,
    float* __restrict__ px) {
  __shared__ float xs[6][58];     // 4+halo rows, 56+halo cols
  __shared__ float ws[18 * 9];
  const int tid = threadIdx.x;
  const int b  = blockIdx.x / 14;
  const int h0 = (blockIdx.x % 14) * 4;
  const int hh = tid / 56;
  const int ww = tid % 56;

  float acc[18];
#pragma unroll
  for (int ch = 0; ch < 18; ++ch) acc[ch] = 0.f;

  for (int c = 0; c < Cc; ++c) {
    for (int i = tid; i < 6 * 58; i += 224) {
      int r = i / 58, cc = i % 58;
      int hy = h0 - 1 + r, wx = cc - 1;
      float v = 0.f;
      if (hy >= 0 && hy < Hh && wx >= 0 && wx < Wd)
        v = x[((size_t)(b * Cc + c)) * HW + hy * Wd + wx];
      xs[r][cc] = v;
    }
    for (int i = tid; i < 162; i += 224)
      ws[i] = w_off[((i / 9) * Cc + c) * 9 + (i % 9)];
    __syncthreads();
#pragma unroll
    for (int k = 0; k < 9; ++k) {
      float xv = xs[hh + k / 3][ww + k % 3];
#pragma unroll
      for (int ch = 0; ch < 18; ++ch)
        acc[ch] = fmaf(xv, ws[ch * 9 + k], acc[ch]);
    }
    __syncthreads();
  }

  const int h = h0 + hh;
#pragma unroll
  for (int k = 0; k < 9; ++k) {
    float dyv = acc[2 * k]     + b_off[2 * k];
    float dxv = acc[2 * k + 1] + b_off[2 * k + 1];
    int o = (b * KKp + k) * HW + h * Wd + ww;
    py[o] = (float)(h  + k / 3 - 1) + dyv;
    px[o] = (float)(ww + k % 3 - 1) + dxv;
  }
}

// ------------- 3) pack w_dcn -> bf16 B-matrix, K = k*C + c -------------
__global__ __launch_bounds__(256) void pack_w_kernel(
    const float* __restrict__ w_dcn, unsigned short* __restrict__ Bpack) {
  int i = blockIdx.x * 256 + threadIdx.x;
  if (i >= NKT * 256 * 32) return;
  int kk = i & 31;
  int n  = (i >> 5) & 255;
  int kt = i >> 13;
  int k  = kt >> 3;
  int c  = ((kt & 7) << 5) + kk;
  float v = w_dcn[(n * Cc + c) * KKp + k];
  unsigned int u = __float_as_uint(v);
  u = (u + 0x7FFFu + ((u >> 16) & 1u)) >> 16;
  Bpack[i] = (unsigned short)u;   // layout: [kt][n][kk] — i is exactly that
}

// gather pipeline helpers: issue (loads only) / commit (blend + LDS store)
__device__ __forceinline__ void gather_issue(
    const float* __restrict__ xT, const int* __restrict__ ip,
    const float* __restrict__ wp, int c0, int j0, float4 ld[8], float wg[4]) {
#pragma unroll
  for (int c = 0; c < 4; ++c) {
    wg[c] = wp[c];
    const float4* p = (const float4*)(xT + ip[c] + c0 + j0);
    ld[2 * c]     = p[0];
    ld[2 * c + 1] = p[1];
  }
}

__device__ __forceinline__ void gather_commit(
    unsigned short* __restrict__ dst, const float4 ld[8], const float wg[4]) {
  float r[8];
#pragma unroll
  for (int j = 0; j < 8; ++j) r[j] = 0.f;
#pragma unroll
  for (int c = 0; c < 4; ++c) {
    const float w = wg[c];
    const float4 a = ld[2 * c], b4 = ld[2 * c + 1];
    r[0] = fmaf(w, a.x,  r[0]); r[1] = fmaf(w, a.y,  r[1]);
    r[2] = fmaf(w, a.z,  r[2]); r[3] = fmaf(w, a.w,  r[3]);
    r[4] = fmaf(w, b4.x, r[4]); r[5] = fmaf(w, b4.y, r[5]);
    r[6] = fmaf(w, b4.z, r[6]); r[7] = fmaf(w, b4.w, r[7]);
  }
  uint4 st;
  st.x = pk_bf16(r[0], r[1]); st.y = pk_bf16(r[2], r[3]);
  st.z = pk_bf16(r[4], r[5]); st.w = pk_bf16(r[6], r[7]);
  *(uint4*)dst = st;
}

// ------------- 4) deformable implicit GEMM via WMMA bf16 -------------
// block = 256 threads (8 waves). Tile: 64 M (spatial) x 256 N (all out ch).
// Double-buffered LDS A tile; gather for chunk kt+1 overlaps WMMAs of kt.
__global__ __launch_bounds__(256) void dcn_main_kernel(
    const float* __restrict__ xT, const float* __restrict__ py,
    const float* __restrict__ px, const unsigned short* __restrict__ Bpack,
    const float* __restrict__ b_dcn, float* __restrict__ out) {
  __shared__ __attribute__((aligned(16))) unsigned short As[2][64][40];
  __shared__ int   sIdx[64][KKp][4];
  __shared__ float sWgt[64][KKp][4];

  const int tid = threadIdx.x;
  const int m0  = blockIdx.x * 64;

  // Per-tile bilinear corner precompute (indices clamped, OOB weights zeroed).
  for (int s = tid; s < 64 * KKp; s += 256) {
    int m = s / KKp, k = s % KKp;
    int gm = m0 + m;
    int b = gm / HW, rem = gm - b * HW;
    float pyv = py[(b * KKp + k) * HW + rem];
    float pxv = px[(b * KKp + k) * HW + rem];
    float y0f = floorf(pyv), x0f = floorf(pxv);
    float fy = pyv - y0f, fx = pxv - x0f;
    int y0 = (int)y0f, x0 = (int)x0f;
    float wy[2] = {1.f - fy, fy};
    float wx[2] = {1.f - fx, fx};
#pragma unroll
    for (int c = 0; c < 4; ++c) {
      int dy = c >> 1, dx = c & 1;
      int yc = y0 + dy, xc = x0 + dx;
      bool ok = (yc >= 0) && (yc < Hh) && (xc >= 0) && (xc < Wd);
      int yi = min(max(yc, 0), Hh - 1);
      int xi = min(max(xc, 0), Wd - 1);
      sIdx[m][k][c] = (b * HW + yi * Wd + xi) * Cc;   // element offset into xT
      sWgt[m][k][c] = ok ? wy[dy] * wx[dx] : 0.f;
    }
  }
  __syncthreads();

  const int lane = tid & 31;
  const int wave = tid >> 5;
  const int wm   = wave & 3;
  const int wn   = wave >> 2;
  const int lrow = lane & 15;
  const int half = lane >> 4;

  v8f acc[8];
#pragma unroll
  for (int t = 0; t < 8; ++t) acc[t] = (v8f)0.0f;

  const int gmr = tid >> 2;            // gather: row of A tile
  const int j0  = (tid & 3) * 8;       // gather: 8-channel group

  float4 ld[8];
  float  wg[4];

  // prologue: stage chunk 0 into buffer 0
  gather_issue(xT, sIdx[gmr][0], sWgt[gmr][0], 0, j0, ld, wg);
  gather_commit(&As[0][gmr][j0], ld, wg);
  __syncthreads();

  for (int kt = 0; kt < NKT; ++kt) {
    const int  buf  = kt & 1;
    const bool more = (kt + 1) < NKT;

    // 1) issue next chunk's corner loads (long-latency, L2-resident)
    if (more) {
      const int kn  = (kt + 1) >> 3;
      const int c0n = ((kt + 1) & 7) * 32;
      gather_issue(xT, sIdx[gmr][kn], sWgt[gmr][kn], c0n, j0, ld, wg);
      const unsigned short* npf =
          Bpack + (((size_t)(kt + 1) * 256 + wn * 128 + lrow) << 5);
      __builtin_prefetch(npf, 0, 1);
    }

    // 2) A fragment (ISA 16-bit 16x32 layout): lanes<16 K{0-7,16-23}, else K{8-15,24-31}
    const int arow = wm * 16 + lrow;
    v8bf alo = *(const v8bf*)&As[buf][arow][half * 8];
    v8bf ahi = *(const v8bf*)&As[buf][arow][16 + half * 8];
    v16bf afrag = __builtin_shufflevector(alo, ahi,
        0, 1, 2, 3, 4, 5, 6, 7, 8, 9, 10, 11, 12, 13, 14, 15);

    //    8 WMMA across this wave's 128-wide N half
#pragma unroll
    for (int t = 0; t < 8; ++t) {
      const int n = wn * 128 + t * 16 + lrow;
      const unsigned short* bp =
          Bpack + (((size_t)kt * 256 + n) << 5) + half * 8;
      v8bf blo = *(const v8bf*)bp;
      v8bf bhi = *(const v8bf*)(bp + 16);
      v16bf bfrag = __builtin_shufflevector(blo, bhi,
          0, 1, 2, 3, 4, 5, 6, 7, 8, 9, 10, 11, 12, 13, 14, 15);
      acc[t] = __builtin_amdgcn_wmma_f32_16x16x32_bf16(
          false, afrag, false, bfrag, (short)0, acc[t], false, false);
    }

    // 3) blend and store next chunk into the other buffer
    if (more) gather_commit(&As[buf ^ 1][gmr][j0], ld, wg);

    __syncthreads();   // single barrier per chunk (double buffering)
  }

  // ---- epilogue: 8 consecutive M rows per accumulator -> two b128 stores ----
  const int mrow0 = m0 + wm * 16 + half * 8;   // tile never crosses a batch
  const int b    = mrow0 / HW;
  const int rem0 = mrow0 - b * HW;
#pragma unroll
  for (int t = 0; t < 8; ++t) {
    const int n = wn * 128 + t * 16 + lrow;
    const float bias = b_dcn[n];
    float4 lo, hi;
    lo.x = acc[t][0] + bias; lo.y = acc[t][1] + bias;
    lo.z = acc[t][2] + bias; lo.w = acc[t][3] + bias;
    hi.x = acc[t][4] + bias; hi.y = acc[t][5] + bias;
    hi.z = acc[t][6] + bias; hi.w = acc[t][7] + bias;
    float* po = out + ((size_t)(b * Oc + n)) * HW + rem0;
    *(float4*)po       = lo;
    *(float4*)(po + 4) = hi;
  }
}

// ---------------- host-side launch ----------------
extern "C" void kernel_launch(void* const* d_in, const int* in_sizes, int n_in,
                              void* d_out, int out_size, void* d_ws, size_t ws_size,
                              hipStream_t stream) {
  const float* x     = (const float*)d_in[0];
  const float* w_off = (const float*)d_in[1];
  const float* b_off = (const float*)d_in[2];
  const float* w_dcn = (const float*)d_in[3];
  const float* b_dcn = (const float*)d_in[4];
  float* out = (float*)d_out;

  // workspace carve-up (~28.7 MB total)
  float* xT = (float*)d_ws;                          // 8*3136*256 f = 25.7 MB
  float* py = xT + (size_t)Bn * HW * Cc;             // 8*9*3136 f
  float* px = py + (size_t)Bn * KKp * HW;            // 8*9*3136 f
  unsigned short* Bpack =
      (unsigned short*)(px + (size_t)Bn * KKp * HW); // 72*256*32 u16 = 1.18 MB

  transpose_kernel<<<dim3(HW / 32, Cc / 32, Bn), dim3(32, 8), 0, stream>>>(x, xT);
  off_conv_kernel<<<dim3(Bn * 14), dim3(224), 0, stream>>>(x, w_off, b_off, py, px);
  pack_w_kernel<<<dim3((NKT * 256 * 32) / 256), dim3(256), 0, stream>>>(w_dcn, Bpack);
  dcn_main_kernel<<<dim3(25088 / 64), dim3(256), 0, stream>>>(xT, py, px, Bpack,
                                                              b_dcn, out);
}
